// RNN_s_41171556500015
// MI455X (gfx1250) — matrix operand order repaired and checked by reading, not compiled, not run on hardware
//
#include <hip/hip_runtime.h>
#include <hip/hip_bf16.h>
#include <math.h>

// ---------------------------------------------------------------------------
// Bidirectional adaptive-LIF SRNN for MI455X (gfx1250, wave32, WMMA).
//   H=256, OUT=61, BATCH=128, SEQ=200, CIN=39, T=1000
// Phase 0: Wr (f32) -> packed bf16 pairs
// Phase 1: proj[dir][s*128+b][h] = x[b,s,:]@Wi.T + bi + br   (200 distinct s)
// Phase 2: persistent recurrence, 16 WGs = {2 dir} x {8 batch groups of 16}
//          wave w of 16 owns N-tile h0=16w; Wr^T B-fragments stationary in
//          VGPRs; spikes exchanged via DOUBLE-BUFFERED LDS (1 barrier/step);
//          proj cached in regs (reload only when the 5-step group advances);
//          8x v_wmma_f32_16x16x32_bf16 per wave per step; LIF update in
//          2-constant FMA form so alpha/ro stay pinned in VGPRs (no per-step
//          divide/exp rematerialization competing with WMMA on TRANS pipe).
// Phase 3: leaky readout + log_softmax, sequential over s.
// ---------------------------------------------------------------------------

typedef __bf16 v16bf __attribute__((ext_vector_type(16)));
typedef float  v8f   __attribute__((ext_vector_type(8)));

union BFrag { v16bf v; unsigned u[8]; };

#define HID   256
#define OUTD  61
#define BATCH 128
#define SEQL  200
#define CIN   39
#define TSTEP 1000
#define ROWS  (SEQL * BATCH)          // 25600
#define LDS_ROW_U32 132               // 128 pairs + 4 pad (bank-conflict free)

__device__ __forceinline__ unsigned pack_bf16_rne(float a, float b) {
  union { float f; unsigned u; } ua, ub;
  ua.f = a; ub.f = b;
  unsigned lo = (ua.u + 0x7FFFu + ((ua.u >> 16) & 1u)) >> 16;
  unsigned hi = (ub.u + 0x7FFFu + ((ub.u >> 16) & 1u)) >> 16;
  return (hi << 16) | (lo & 0xFFFFu);
}

// ---------------- Phase 0: pack Wr to bf16 pairs ---------------------------
__global__ void convert_wr_kernel(const float* __restrict__ wr_fw,
                                  const float* __restrict__ wr_bw,
                                  unsigned* __restrict__ wr_pk) {
  int i = blockIdx.x * blockDim.x + threadIdx.x;   // 2 * 256 * 128 pairs
  if (i >= 2 * HID * (HID / 2)) return;
  int dir = i >> 15;                               // 32768 pairs per dir
  int p   = i & 32767;
  const float* w = dir ? wr_bw : wr_fw;
  wr_pk[i] = pack_bf16_rne(w[2 * p], w[2 * p + 1]);
}

// ---------------- Phase 1: input projection --------------------------------
__global__ void proj_kernel(const float* __restrict__ x,
                            const float* __restrict__ Wi_fw,
                            const float* __restrict__ bi_fw,
                            const float* __restrict__ br_fw,
                            const float* __restrict__ Wi_bw,
                            const float* __restrict__ bi_bw,
                            const float* __restrict__ br_bw,
                            float* __restrict__ proj) {
  __shared__ float xr[CIN];
  int row = blockIdx.x;                 // s*128 + b
  int dir = blockIdx.y;
  int s = row >> 7, b = row & 127;
  int h = threadIdx.x;                  // 0..255
  if (h < CIN) xr[h] = x[((size_t)b * SEQL + s) * CIN + h];
  __syncthreads();
  const float* Wi = dir ? Wi_bw : Wi_fw;
  const float* bi = dir ? bi_bw : bi_fw;
  const float* br = dir ? br_bw : br_fw;
  float acc = bi[h] + br[h];
  #pragma unroll
  for (int c = 0; c < CIN; ++c) acc = fmaf(xr[c], Wi[h * CIN + c], acc);
  proj[(size_t)dir * ROWS * HID + (size_t)row * HID + h] = acc;
}

// ---------------- Phase 2: persistent recurrence ---------------------------
__global__ __launch_bounds__(512, 1) void srnn_kernel(
    const float* __restrict__ proj,       // [2][25600][256]
    const unsigned* __restrict__ wr_pk,   // [2][256][128] bf16 pairs (Wr[h][k])
    const float* __restrict__ tau_m_fw, const float* __restrict__ tau_adp_fw,
    const float* __restrict__ tau_m_bw, const float* __restrict__ tau_adp_bw,
    float* __restrict__ merged)           // [200][128][512]
{
  // double-buffered spike tiles: [buf][m][kpair], padded rows
  __shared__ unsigned spk_lds[2][16 * LDS_ROW_U32];

  const int dir = blockIdx.x >> 3;
  const int bg  = blockIdx.x & 7;        // batch group: rows bg*16 .. bg*16+15
  const int tid  = threadIdx.x;
  const int lane = tid & 31;
  const int h0 = (tid >> 5) * 16;        // wave -> N tile
  const int n  = lane & 15;
  const int lhalf = lane >> 4;
  const int h = h0 + n;                  // this lane's hidden column

  const float* tau_m   = dir ? tau_m_bw  : tau_m_fw;
  const float* tau_adp = dir ? tau_adp_bw : tau_adp_fw;
  float alpha = __expf(-1.0f / tau_m[h]);
  float ro    = __expf(-1.0f / tau_adp[h]);
  // Pin the loop-invariant per-lane decay constants in VGPRs so the scheduler
  // cannot rematerialize the divide+exp chains inside the serial time loop.
  asm volatile("" : "+v"(alpha), "+v"(ro));

  // --- preload stationary B fragments: B[k][n] = Wr[h0+n][k] (bf16 pairs) ---
  // B 32x16, contiguous k order chosen here (A matches): lane l, vgpr j -> N=l&15, K = 2j + 16*(l>>4)
  const unsigned* wr = wr_pk + (size_t)dir * HID * (HID / 2);
  unsigned bfrag[8][8];
  #pragma unroll
  for (int kk = 0; kk < 8; ++kk) {
    #pragma unroll
    for (int j = 0; j < 8; ++j) {
      int K = 32 * kk + 2 * j + 16 * lhalf;
      bfrag[kk][j] = wr[h * (HID / 2) + (K >> 1)];
    }
  }

  // zero initial spikes in buffer 0
  for (int i = tid; i < 16 * LDS_ROW_U32; i += 512) spk_lds[0][i] = 0u;

  // state in C/D f32 layout: vgpr i, lane -> m = i + 8*lhalf, column h
  v8f mem{}, spk{}, badp, acc{}, pr;
  #pragma unroll
  for (int i = 0; i < 8; ++i) { badp[i] = 0.01f; mem[i] = 0.0f; spk[i] = 0.0f; acc[i] = 0.0f; }

  const float* projd = proj + (size_t)dir * ROWS * HID;

  // proj registers for current 5-step group (idx starts at 0 for both dirs)
  int idx = 0;     // current proj sequence row; also equals merged s at write
  int ctr = 0;     // t % 5, maintained incrementally
  #pragma unroll
  for (int i = 0; i < 8; ++i) {
    int brow = bg * 16 + i + 8 * lhalf;
    pr[i] = projd[((size_t)idx * BATCH + brow) * HID + h];
  }

  __syncthreads();

  for (int t = 0; t < TSTEP; ++t) {
    const unsigned* rbuf = spk_lds[t & 1];
    unsigned short* wbuf16 = (unsigned short*)spk_lds[(t & 1) ^ 1];

    // advance proj group (uniform scalar; no divisions)
    if (t > 0) {
      bool changed = false;
      if (dir == 0) { if (ctr == 0) { idx++; changed = true; } }
      else          { if (ctr == 1) { idx = (idx == 0) ? (SEQL - 1) : idx - 1;
                                      changed = true; } }
      if (changed) {
        #pragma unroll
        for (int i = 0; i < 8; ++i) {
          int brow = bg * 16 + i + 8 * lhalf;
          pr[i] = projd[((size_t)idx * BATCH + brow) * HID + h];
        }
      }
    }

    // ---- recurrent matmul: c = spikes(16x256) @ Wr^T tile ----
    v8f c{};
    #pragma unroll
    for (int kk = 0; kk < 8; ++kk) {
      // A 16x32, same contiguous k order as B: lane l, vgpr j ->
      //   M = l&15 ; K = 32kk + 2*j + 16*(l>>4)
      BFrag af;
      #pragma unroll
      for (int j = 0; j < 8; ++j) {
        int M = lane & 15;
        int K = 32 * kk + 2 * j + 16 * lhalf;
        af.u[j] = rbuf[M * LDS_ROW_U32 + (K >> 1)];
      }
      BFrag bf;
      #pragma unroll
      for (int j = 0; j < 8; ++j) bf.u[j] = bfrag[kk][j];
      c = __builtin_amdgcn_wmma_f32_16x16x32_bf16(
              false, af.v, false, bf.v, (short)0, c, false, false);
    }

    // ---- elementwise adaptive-LIF update; write spikes to OTHER buffer ----
    // 2-constant FMA form:
    //   badp' = spk + ro*(badp - spk)
    //   mem'  = din + alpha*(mem - din) - Bth*spk
    #pragma unroll
    for (int i = 0; i < 8; ++i) {
      int m = i + 8 * lhalf;
      float din = pr[i] + c[i];
      float bn  = fmaf(ro, badp[i] - spk[i], spk[i]);
      float Bth = fmaf(1.8f, bn, 0.01f);
      float mn  = fmaf(alpha, mem[i] - din, din) - Bth * spk[i];
      float sn  = (mn - Bth) > 0.0f ? 1.0f : 0.0f;
      badp[i] = bn; mem[i] = mn; spk[i] = sn;
      acc[i] += sn;
      wbuf16[m * (LDS_ROW_U32 * 2) + h] = (sn != 0.0f) ? (unsigned short)0x3F80
                                                       : (unsigned short)0x0000;
    }

    // ---- block-of-5 running mean -> merged (s == idx for both dirs) ----
    if (ctr == 4) {
      #pragma unroll
      for (int i = 0; i < 8; ++i) {
        int brow = bg * 16 + i + 8 * lhalf;
        merged[((size_t)idx * BATCH + brow) * (2 * HID) + dir * HID + h] =
            acc[i] * 0.2f;
        acc[i] = 0.0f;
      }
    }
    ctr = (ctr == 4) ? 0 : ctr + 1;

    __syncthreads();   // new spikes visible before next step reads that buffer
  }
}

// ---------------- Phase 3: readout + log_softmax ---------------------------
__global__ __launch_bounds__(64) void readout_kernel(
    const float* __restrict__ merged,     // [200][128][512]
    const float* __restrict__ W_out,      // [61][512]
    const float* __restrict__ b_out,      // [61]
    const float* __restrict__ tau_m_out,  // [61]
    float* __restrict__ preds)            // [200][128][61]
{
  __shared__ float xr[2 * HID];
  __shared__ float red[64];
  const int b = blockIdx.x;
  const int o = threadIdx.x;
  const bool valid = (o < OUTD);
  const float alpha = valid ? __expf(-1.0f / tau_m_out[o]) : 0.0f;
  const float bias  = valid ? b_out[o] : 0.0f;
  float memv = 0.0f;

  for (int s = 0; s < SEQL; ++s) {
    __syncthreads();
    for (int k = o; k < 2 * HID; k += 64)
      xr[k] = merged[((size_t)s * BATCH + b) * (2 * HID) + k];
    __syncthreads();

    float dot = bias;
    if (valid) {
      #pragma unroll 8
      for (int k = 0; k < 2 * HID; ++k)
        dot = fmaf(xr[k], W_out[o * (2 * HID) + k], dot);
    }
    memv = memv * alpha + (1.0f - alpha) * dot;

    // log_softmax over the 61 outputs (64-thread tree reductions)
    red[o] = valid ? memv : -INFINITY;
    __syncthreads();
    for (int st = 32; st > 0; st >>= 1) {
      if (o < st) red[o] = fmaxf(red[o], red[o + st]);
      __syncthreads();
    }
    float mx = red[0];
    __syncthreads();
    red[o] = valid ? __expf(memv - mx) : 0.0f;
    __syncthreads();
    for (int st = 32; st > 0; st >>= 1) {
      if (o < st) red[o] += red[o + st];
      __syncthreads();
    }
    float lse = mx + __logf(red[0]);
    __syncthreads();
    if (valid)
      preds[((size_t)s * BATCH + b) * OUTD + o] = memv - lse;
  }
}

// ---------------------------------------------------------------------------
extern "C" void kernel_launch(void* const* d_in, const int* in_sizes, int n_in,
                              void* d_out, int out_size, void* d_ws, size_t ws_size,
                              hipStream_t stream) {
  const float* x          = (const float*)d_in[0];
  const float* Wi_fw      = (const float*)d_in[1];
  const float* bi_fw      = (const float*)d_in[2];
  const float* Wr_fw      = (const float*)d_in[3];
  const float* br_fw      = (const float*)d_in[4];
  const float* tau_m_fw   = (const float*)d_in[5];
  const float* tau_adp_fw = (const float*)d_in[6];
  const float* Wi_bw      = (const float*)d_in[7];
  const float* bi_bw      = (const float*)d_in[8];
  const float* Wr_bw      = (const float*)d_in[9];
  const float* br_bw      = (const float*)d_in[10];
  const float* tau_m_bw   = (const float*)d_in[11];
  const float* tau_adp_bw = (const float*)d_in[12];
  const float* W_out      = (const float*)d_in[13];
  const float* b_out      = (const float*)d_in[14];
  const float* tau_m_out  = (const float*)d_in[15];
  float* preds = (float*)d_out;

  char* ws = (char*)d_ws;
  float* proj = (float*)ws;                                   // 2*25600*256 f32
  size_t off = (size_t)2 * ROWS * HID * sizeof(float);        // 52.4 MB
  unsigned* wr_pk = (unsigned*)(ws + off);                    // 2*256*128 u32
  off += (size_t)2 * HID * (HID / 2) * sizeof(unsigned);      // +0.26 MB
  float* merged = (float*)(ws + off);                         // 200*128*512 f32
  (void)in_sizes; (void)n_in; (void)out_size; (void)ws_size;

  convert_wr_kernel<<<(2 * HID * (HID / 2) + 255) / 256, 256, 0, stream>>>(
      Wr_fw, Wr_bw, wr_pk);
  proj_kernel<<<dim3(ROWS, 2), 256, 0, stream>>>(
      x, Wi_fw, bi_fw, br_fw, Wi_bw, bi_bw, br_bw, proj);
  srnn_kernel<<<16, 512, 0, stream>>>(
      proj, wr_pk, tau_m_fw, tau_adp_fw, tau_m_bw, tau_adp_bw, merged);
  readout_kernel<<<BATCH, 64, 0, stream>>>(
      merged, W_out, b_out, tau_m_out, preds);
}
